// JointAttention_47614007443765
// MI455X (gfx1250) — matrix operand order repaired and checked by reading, not compiled, not run on hardware
//
#include <hip/hip_runtime.h>
#include <hip/hip_bf16.h>

typedef __attribute__((ext_vector_type(16))) _Float16 v16h;
typedef __attribute__((ext_vector_type(8)))  float    v8f;

#define CH    512
#define TI    4096
#define TS    1024
#define NB    2
#define NHEAD 8
#define HD    64

static __device__ __forceinline__ int lane_id() { return (int)(threadIdx.x & 31u); }

static __device__ __forceinline__ v8f wmma16(v16h a, v16h b, v8f c) {
  return __builtin_amdgcn_wmma_f32_16x16x32_f16(false, a, false, b, (short)0, c, false, false);
}

// A fragment: 16(M) x 32(K) from row-major source, leading dim ld (elements).
// Lanes 0-15: row m=lane, K = {0..7, 16..23}; lanes 16-31: row m=lane-16, K = {8..15, 24..31}.
static __device__ __forceinline__ v16h ld_fragA(const _Float16* __restrict__ src, int ld) {
  int lane = lane_id();
  int m  = lane & 15;
  int kb = (lane >> 4) << 3;           // 0 or 8
  const _Float16* p = src + (size_t)m * ld + kb;
  v16h f;
#pragma unroll
  for (int i = 0; i < 8; ++i) f[i] = p[i];
#pragma unroll
  for (int i = 0; i < 8; ++i) f[8 + i] = p[16 + i];
  return f;
}

// B fragment: 32(K) x 16(N) where column n of B is row n of row-major matrix y
// (i.e. computing X @ Y^T). Lane owns column n = lane&15; lanes<16 hold K={0..7,16..23},
// lanes>=16 hold K={8..15,24..31} (same k order as the A fragment).
static __device__ __forceinline__ v16h ld_fragB_rows(const _Float16* __restrict__ y, int ld) {
  int lane = lane_id();
  int n  = lane & 15;
  int k0 = (lane >> 4) << 3;           // 0 or 8
  const _Float16* p = y + (size_t)n * ld + k0;
  v16h f;
#pragma unroll
  for (int i = 0; i < 8; ++i) f[i] = p[i];
#pragma unroll
  for (int i = 0; i < 8; ++i) f[8 + i] = p[16 + i];
  return f;
}

// -------------------- GroupNorm --------------------
__global__ void gn_stats(const float* __restrict__ src, int T,
                         float* __restrict__ mu, float* __restrict__ rs) {
  int bg = blockIdx.x;                  // b*32 + g
  int b = bg >> 5, g = bg & 31;
  const float* p = src + (size_t)b * T * CH + g * 16;
  int n = T * 16;
  float s = 0.f, s2 = 0.f;
  for (int i = threadIdx.x; i < n; i += blockDim.x) {
    int t = i >> 4, c = i & 15;
    float v = p[(size_t)t * CH + c];
    s += v; s2 += v * v;
  }
  __shared__ float sh[256], sh2[256];
  sh[threadIdx.x] = s; sh2[threadIdx.x] = s2;
  __syncthreads();
  for (int off = blockDim.x >> 1; off > 0; off >>= 1) {
    if ((int)threadIdx.x < off) { sh[threadIdx.x] += sh[threadIdx.x + off]; sh2[threadIdx.x] += sh2[threadIdx.x + off]; }
    __syncthreads();
  }
  if (threadIdx.x == 0) {
    float m = sh[0] / (float)n;
    float v = sh2[0] / (float)n - m * m;
    mu[bg] = m;
    rs[bg] = rsqrtf(v + 1e-5f);
  }
}

__global__ void gn_apply(const float* __restrict__ src, const float* __restrict__ mu,
                         const float* __restrict__ rs, const float* __restrict__ gamma,
                         const float* __restrict__ beta, _Float16* __restrict__ dst, int T) {
  size_t total = (size_t)NB * T * CH;
  for (size_t i = (size_t)blockIdx.x * blockDim.x + threadIdx.x; i < total;
       i += (size_t)gridDim.x * blockDim.x) {
    int c  = (int)(i % CH);
    int bt = (int)(i / CH);
    int b  = bt / T;
    int g  = c >> 4;
    float v = (src[i] - mu[b * 32 + g]) * rs[b * 32 + g] * gamma[c] + beta[c];
    dst[i] = (_Float16)v;
  }
}

// -------------------- Weight repack (f32 -> f16) --------------------
__global__ void pack_weights(const float* __restrict__ xqkv_w, const float* __restrict__ jqkv_w,
                             const float* __restrict__ proj_w,
                             _Float16* __restrict__ wq, _Float16* __restrict__ wk,
                             _Float16* __restrict__ wv, _Float16* __restrict__ wp) {
  const int NW = CH * CH;               // 262144
  const int NP = CH * 3 * CH;           // 786432
  int total = 3 * NW + NP;
  for (int i = blockIdx.x * blockDim.x + threadIdx.x; i < total; i += gridDim.x * blockDim.x) {
    if (i < NW)            wq[i] = (_Float16)xqkv_w[i];
    else if (i < 2 * NW) { int j = i - NW;     wk[j] = (_Float16)jqkv_w[NW + j]; }
    else if (i < 3 * NW) { int j = i - 2 * NW; wv[j] = (_Float16)jqkv_w[2 * NW + j]; }
    else {
      int j = i - 3 * NW;               // wp layout: [o][k*512 + c]
      int o = j / (3 * CH);
      int r = j - o * (3 * CH);
      int k = r >> 9;
      int c = r & 511;
      wp[j] = (_Float16)proj_w[((size_t)o * CH + c) * 3 + k];
    }
  }
}

// -------------------- projection GEMM: out = scale*(A @ W^T + bias) --------------------
// Each wave computes a 16x64 strip: 4 accumulators share one A fragment per k-step.
// Fully unrolled k-loop -> one basic block; the scheduler hoists each step's load
// clauses above earlier WMMAs (software pipelining without register-rotation movs).
// transposed==0: out is [rows, CH] f16.  transposed==1: out is [B, CH, TS] f16 (for V^T).
static __device__ __forceinline__ void store_qkv_tile(v8f acc, _Float16* __restrict__ out,
                                                      const float* __restrict__ bias,
                                                      int mt, int nt, float scale, int transposed) {
  int lane = lane_id();
  int n = lane & 15, mh = (lane >> 4) << 3;
  int col = nt * 16 + n;
  float bv = bias[col];
  if (!transposed) {
    _Float16* o = out + ((size_t)mt * 16 + mh) * CH + col;
#pragma unroll
    for (int r = 0; r < 8; ++r) o[(size_t)r * CH] = (_Float16)(scale * (acc[r] + bv));
  } else {
#pragma unroll
    for (int r = 0; r < 8; ++r) {
      int row = mt * 16 + mh + r;                 // global row over B*TS
      out[((size_t)(row >> 10) * CH + col) * TS + (row & (TS - 1))] =
          (_Float16)(scale * (acc[r] + bv));
    }
  }
}

__global__ __launch_bounds__(128) void gemm_qkv(const _Float16* __restrict__ A,
                                                const _Float16* __restrict__ W,
                                                const float* __restrict__ bias,
                                                _Float16* __restrict__ out, float scale,
                                                int transposed) {
  int mt = blockIdx.x;                                    // 16-row tile (rows = B*T merged)
  int nt0 = (blockIdx.y * 4 + (threadIdx.x >> 5)) * 4;    // 4 consecutive 16-col tiles
  const _Float16* a0 = A + (size_t)mt * 16 * CH;
  const _Float16* w0 = W + (size_t)nt0 * 16 * CH;
  v8f acc0 = {}, acc1 = {}, acc2 = {}, acc3 = {};
#pragma unroll
  for (int k = 0; k < CH; k += 32) {
    v16h af = ld_fragA(a0 + k, CH);
    v16h b0 = ld_fragB_rows(w0 + k, CH);
    v16h b1 = ld_fragB_rows(w0 + (size_t)16 * CH + k, CH);
    v16h b2 = ld_fragB_rows(w0 + (size_t)32 * CH + k, CH);
    v16h b3 = ld_fragB_rows(w0 + (size_t)48 * CH + k, CH);
    acc0 = wmma16(af, b0, acc0);
    acc1 = wmma16(af, b1, acc1);
    acc2 = wmma16(af, b2, acc2);
    acc3 = wmma16(af, b3, acc3);
  }
  store_qkv_tile(acc0, out, bias, mt, nt0 + 0, scale, transposed);
  store_qkv_tile(acc1, out, bias, mt, nt0 + 1, scale, transposed);
  store_qkv_tile(acc2, out, bias, mt, nt0 + 2, scale, transposed);
  store_qkv_tile(acc3, out, bias, mt, nt0 + 3, scale, transposed);
}

// -------------------- Cross-attention: 16 query rows per block, 4 waves --------------------
__global__ __launch_bounds__(128) void attn_kernel(const _Float16* __restrict__ Q,
                                                   const _Float16* __restrict__ K,
                                                   const _Float16* __restrict__ Vt,
                                                   _Float16* __restrict__ att) {
  __shared__ float S[16 * TS];          // 64 KB score tile
  int id = blockIdx.x;                  // (b*NHEAD + h)*(TI/16) + mt
  int mt = id & 255; id >>= 8;
  int h  = id & 7;
  int b  = id >> 3;
  int wave = threadIdx.x >> 5;
  int lane = lane_id();

  const _Float16* Qb  = Q + ((size_t)b * TI + (size_t)mt * 16) * CH + h * HD;
  const _Float16* Kb  = K + (size_t)b * TS * CH + h * HD;
  const _Float16* Vtb = Vt + ((size_t)b * CH + h * HD + wave * 16) * TS;

  // ---- Phase 1: S[m][s] = (scaled Q) . (scaled K), wave w covers s in [w*256, w*256+256) ----
  v16h qf0 = ld_fragA(Qb, CH);
  v16h qf1 = ld_fragA(Qb + 32, CH);
#pragma unroll
  for (int i = 0; i < 16; ++i) {
    int nt = wave * 16 + i;
    const _Float16* kp = Kb + (size_t)nt * 16 * CH;
    v16h kb0 = ld_fragB_rows(kp, CH);
    v16h kb1 = ld_fragB_rows(kp + 32, CH);
    v8f acc = {};
    acc = wmma16(qf0, kb0, acc);
    acc = wmma16(qf1, kb1, acc);
    int n = lane & 15, mh = (lane >> 4) << 3;
    float* sp = S + (size_t)mh * TS + nt * 16 + n;
#pragma unroll
    for (int r = 0; r < 8; ++r) sp[(size_t)r * TS] = acc[r];
  }
  __syncthreads();

  // ---- Phase 2: f32 softmax, normalized in place (8 threads per row) ----
  {
    int row = threadIdx.x >> 3;
    int sub = threadIdx.x & 7;
    float* rp = S + (size_t)row * TS;
    float mx = -1e30f;
    for (int j = sub; j < TS; j += 8) mx = fmaxf(mx, rp[j]);
    mx = fmaxf(mx, __shfl_xor(mx, 1, 8));
    mx = fmaxf(mx, __shfl_xor(mx, 2, 8));
    mx = fmaxf(mx, __shfl_xor(mx, 4, 8));
    float sum = 0.f;
    for (int j = sub; j < TS; j += 8) { float e = __expf(rp[j] - mx); rp[j] = e; sum += e; }
    sum += __shfl_xor(sum, 1, 8);
    sum += __shfl_xor(sum, 2, 8);
    sum += __shfl_xor(sum, 4, 8);
    float inv = 1.0f / sum;
    for (int j = sub; j < TS; j += 8) rp[j] *= inv;
  }
  __syncthreads();

  // ---- Phase 3: O = P @ V; wave w handles head-dim cols [w*16, w*16+16).
  //      V is pre-transposed ([B,C,Ts]) so B-fragment lanes read contiguous 16B runs. ----
  int m = lane & 15, kb = (lane >> 4) << 3;
  v8f acc = {};
#pragma unroll 8
  for (int k0 = 0; k0 < TS; k0 += 32) {
    const float* p = S + (size_t)m * TS + k0 + kb;
    v16h af;
#pragma unroll
    for (int i = 0; i < 8; ++i) af[i] = (_Float16)p[i];
#pragma unroll
    for (int i = 0; i < 8; ++i) af[8 + i] = (_Float16)p[16 + i];
    v16h bf = ld_fragB_rows(Vtb + k0, TS);
    acc = wmma16(af, bf, acc);
  }
  int n = lane & 15, mh = (lane >> 4) << 3;
  _Float16* o = att + ((size_t)b * TI + (size_t)mt * 16 + mh) * CH + h * HD + wave * 16 + n;
#pragma unroll
  for (int r = 0; r < 8; ++r) o[(size_t)r * CH] = (_Float16)acc[r];
}

// -------------------- conv3(SAME) as K=1536 GEMM + residual, writes [B,C,Ti] f32 --------------------
static __device__ __forceinline__ v16h ld_fragA_conv(const _Float16* __restrict__ attb,
                                                     int tbase, int c0) {
  int lane = lane_id();
  int m  = lane & 15;
  int kb = (lane >> 4) << 3;
  int t  = tbase + m;
  v16h f;
  if ((unsigned)t < (unsigned)TI) {
    const _Float16* p = attb + (size_t)t * CH + c0 + kb;
#pragma unroll
    for (int i = 0; i < 8; ++i) f[i] = p[i];
#pragma unroll
    for (int i = 0; i < 8; ++i) f[8 + i] = p[16 + i];
  } else {
#pragma unroll
    for (int i = 0; i < 16; ++i) f[i] = (_Float16)0.f;
  }
  return f;
}

static __device__ __forceinline__ void store_conv_tile(v8f acc, float* __restrict__ out,
                                                       const float* __restrict__ pb,
                                                       const float* __restrict__ image,
                                                       int b, int mt, int nt) {
  int lane = lane_id();
  int n = lane & 15, mh = (lane >> 4) << 3;
  int c = nt * 16 + n;
  float bv = pb[c];
#pragma unroll
  for (int r = 0; r < 8; ++r) {
    int t = mt * 16 + mh + r;
    out[((size_t)b * CH + c) * TI + t] =
        acc[r] + bv + image[((size_t)b * TI + t) * CH + c];
  }
}

__global__ __launch_bounds__(128) void conv3_residual(const _Float16* __restrict__ att,
                                                      const _Float16* __restrict__ wp,
                                                      const float* __restrict__ pb,
                                                      const float* __restrict__ image,
                                                      float* __restrict__ out) {
  int id = blockIdx.x;                  // b*(TI/16) + mt
  int mt = id & 255;
  int b  = id >> 8;
  int nt0 = (blockIdx.y * 4 + (threadIdx.x >> 5)) * 4;    // 4 consecutive 16-col tiles
  const int LDW = 3 * CH;
  const _Float16* ab = att + (size_t)b * TI * CH;
  const _Float16* w0 = wp + (size_t)nt0 * 16 * LDW;
  v8f acc0 = {}, acc1 = {}, acc2 = {}, acc3 = {};
#pragma unroll 8
  for (int kk = 0; kk < LDW; kk += 32) {
    int shift = kk >> 9;                // 0,1,2 -> taps t-1, t, t+1
    int c0    = kk & 511;
    v16h af = ld_fragA_conv(ab, mt * 16 + shift - 1, c0);
    v16h b0 = ld_fragB_rows(w0 + kk, LDW);
    v16h b1 = ld_fragB_rows(w0 + (size_t)16 * LDW + kk, LDW);
    v16h b2 = ld_fragB_rows(w0 + (size_t)32 * LDW + kk, LDW);
    v16h b3 = ld_fragB_rows(w0 + (size_t)48 * LDW + kk, LDW);
    acc0 = wmma16(af, b0, acc0);
    acc1 = wmma16(af, b1, acc1);
    acc2 = wmma16(af, b2, acc2);
    acc3 = wmma16(af, b3, acc3);
  }
  store_conv_tile(acc0, out, pb, image, b, mt, nt0 + 0);
  store_conv_tile(acc1, out, pb, image, b, mt, nt0 + 1);
  store_conv_tile(acc2, out, pb, image, b, mt, nt0 + 2);
  store_conv_tile(acc3, out, pb, image, b, mt, nt0 + 3);
}

// -------------------- host --------------------
extern "C" void kernel_launch(void* const* d_in, const int* in_sizes, int n_in,
                              void* d_out, int out_size, void* d_ws, size_t ws_size,
                              hipStream_t stream) {
  const float* image  = (const float*)d_in[0];
  const float* st     = (const float*)d_in[1];
  const float* xn_w   = (const float*)d_in[2];
  const float* xn_b   = (const float*)d_in[3];
  const float* jn_w   = (const float*)d_in[4];
  const float* jn_b   = (const float*)d_in[5];
  const float* xqkv_w = (const float*)d_in[6];
  const float* xqkv_b = (const float*)d_in[7];
  const float* jqkv_w = (const float*)d_in[8];
  const float* jqkv_b = (const float*)d_in[9];
  const float* proj_w = (const float*)d_in[10];
  const float* proj_b = (const float*)d_in[11];
  float* out = (float*)d_out;

  char* ws = (char*)d_ws;
  size_t off = 0;
  auto alloc = [&](size_t bytes) -> void* {
    off = (off + 255) & ~(size_t)255;
    void* p = ws + off;
    off += bytes;
    return p;
  };
  float*    mu_x = (float*)alloc(64 * sizeof(float));
  float*    rs_x = (float*)alloc(64 * sizeof(float));
  float*    mu_j = (float*)alloc(64 * sizeof(float));
  float*    rs_j = (float*)alloc(64 * sizeof(float));
  _Float16* xh   = (_Float16*)alloc((size_t)NB * TI * CH * 2);
  _Float16* jh   = (_Float16*)alloc((size_t)NB * TS * CH * 2);
  _Float16* wq   = (_Float16*)alloc((size_t)CH * CH * 2);
  _Float16* wk   = (_Float16*)alloc((size_t)CH * CH * 2);
  _Float16* wv   = (_Float16*)alloc((size_t)CH * CH * 2);
  _Float16* wp   = (_Float16*)alloc((size_t)CH * 3 * CH * 2);
  _Float16* Qs   = (_Float16*)alloc((size_t)NB * TI * CH * 2);
  _Float16* Ks   = (_Float16*)alloc((size_t)NB * TS * CH * 2);
  _Float16* Vt   = (_Float16*)alloc((size_t)NB * CH * TS * 2);
  _Float16* attb = (_Float16*)alloc((size_t)NB * TI * CH * 2);
  (void)ws_size; (void)in_sizes; (void)n_in; (void)out_size;

  gn_stats<<<NB * 32, 256, 0, stream>>>(image, TI, mu_x, rs_x);
  gn_stats<<<NB * 32, 256, 0, stream>>>(st, TS, mu_j, rs_j);
  gn_apply<<<1024, 256, 0, stream>>>(image, mu_x, rs_x, xn_w, xn_b, xh, TI);
  gn_apply<<<512, 256, 0, stream>>>(st, mu_j, rs_j, jn_w, jn_b, jh, TS);
  pack_weights<<<1024, 256, 0, stream>>>(xqkv_w, jqkv_w, proj_w, wq, wk, wv, wp);

  const float scale = 0.35355339059327373f;   // 1 / sqrt(sqrt(64))
  gemm_qkv<<<dim3(NB * TI / 16, 2), 128, 0, stream>>>(xh, wq, xqkv_b, Qs, scale, 0);
  gemm_qkv<<<dim3(NB * TS / 16, 2), 128, 0, stream>>>(jh, wk, jqkv_b + CH, Ks, scale, 0);
  gemm_qkv<<<dim3(NB * TS / 16, 2), 128, 0, stream>>>(jh, wv, jqkv_b + 2 * CH, Vt, 1.0f, 1);

  attn_kernel<<<NB * NHEAD * (TI / 16), 128, 0, stream>>>(Qs, Ks, Vt, attb);

  conv3_residual<<<dim3(NB * (TI / 16), 2), 128, 0, stream>>>(attb, wp, proj_b, image, out);
}
